// LearnableRelMultiHeadAttn_44787918962705
// MI455X (gfx1250) — matrix operand order repaired and checked by reading, not compiled, not run on hardware
//
#include <hip/hip_runtime.h>

typedef __attribute__((ext_vector_type(16))) __bf16 v16bf;
typedef __attribute__((ext_vector_type(8)))  float  v8f;

#define T_LEN 1024
#define BATCH 4
#define NH    16
#define DH    64
#define DM    1024
#define QSCALE 0.125f

union FragAB { v16bf v; uint2 u2[4]; };

static __device__ __forceinline__ v8f wmma(v16bf a, v16bf b, v8f c) {
  return __builtin_amdgcn_wmma_f32_16x16x32_bf16(false, a, false, b, (short)0, c, false, false);
}

// A-frag (16xK tile chunk at kc, K=32): lane<16 -> row=lane, K = kc+{0..7,16..23};
// lane>=16 -> row=lane-16, K = kc+{8..15,24..31}.
static __device__ __forceinline__ v16bf fragA(const __bf16* base, int stride, int rowBase, int kc) {
  int lane = threadIdx.x & 31;
  int hf = lane >> 4, lr = lane & 15;
  const __bf16* p = base + (rowBase + lr) * stride + kc + hf * 8;
  FragAB f;
  f.u2[0] = *(const uint2*)(p);
  f.u2[1] = *(const uint2*)(p + 4);
  f.u2[2] = *(const uint2*)(p + 16);
  f.u2[3] = *(const uint2*)(p + 20);
  return f.v;
}

// B-frag (32x16 chunk): lane<16 -> col=n0+lane, K = kc+{0..7,16..23};
// lane>=16 -> col=n0+lane-16, K = kc+{8..15,24..31}.
static __device__ __forceinline__ v16bf fragB(const __bf16* base, int stride, int kc, int n0) {
  int lane = threadIdx.x & 31;
  int hf = lane >> 4, lr = lane & 15;
  const __bf16* p = base + (kc + hf * 8) * stride + n0 + lr;
  v16bf b;
#pragma unroll
  for (int i = 0; i < 8; ++i) {
    b[i] = p[i * stride];
    b[8 + i] = p[(16 + i) * stride];
  }
  return b;
}

// gfx1250 async copy: global -> LDS, 16B per lane, tracked by ASYNCcnt.
static __device__ __forceinline__ void async_b128(unsigned ldsOff, const __bf16* g) {
  asm volatile("global_load_async_to_lds_b128 %0, %1, off" :: "v"(ldsOff), "v"(g) : "memory");
}
static __device__ __forceinline__ void wait_async0() {
#if __has_builtin(__builtin_amdgcn_s_wait_asynccnt)
  __builtin_amdgcn_s_wait_asynccnt(0);
#else
  asm volatile("s_wait_asynccnt 0x0" ::: "memory");
#endif
}

// ---------------- Kernel 1: QKV projection GEMM ----------------
// heads[m][n] = sum_k w[m][k] * Wqkv[n][k], m = t*B+b (4096), n (3072), k (1024)
// Q -> [B,H,T,64] (scaled), K -> [B,H,64,T] (pre-transposed for attention), V -> [B,H,T,64]
__global__ __launch_bounds__(256) void k_qkv(const float* __restrict__ w,
                                             const float* __restrict__ Wqkv,
                                             __bf16* __restrict__ Qb,
                                             __bf16* __restrict__ Kt,
                                             __bf16* __restrict__ Vb) {
  const int AS = 40, BS = 68;
  __shared__ __bf16 Asm[64 * 40];
  __shared__ __bf16 Bsm[32 * 68];
  int tid = threadIdx.x, lane = tid & 31, wv = tid >> 5;
  int wr = wv >> 1, wc = wv & 1;
  int mBase = blockIdx.x * 64, nBase = blockIdx.y * 64;
  v8f acc0 = {}, acc1 = {};
  for (int kb = 0; kb < DM; kb += 32) {
    __syncthreads();
#pragma unroll
    for (int i = 0; i < 8; ++i) {
      int e = tid + 256 * i;
      int m = e >> 5, k = e & 31;
      Asm[m * AS + k] = (__bf16)w[(mBase + m) * DM + kb + k];
    }
#pragma unroll
    for (int i = 0; i < 8; ++i) {
      int e = tid + 256 * i;
      int n = e >> 5, k = e & 31;
      Bsm[k * BS + n] = (__bf16)Wqkv[(long)(nBase + n) * DM + kb + k];
    }
    if (kb + 32 < DM) {
      __builtin_prefetch(&Wqkv[(long)(nBase + (tid & 63)) * DM + kb + 32], 0, 3);
      __builtin_prefetch(&w[(mBase + (tid & 63)) * DM + kb + 32], 0, 3);
    }
    __syncthreads();
    v16bf a = fragA(Asm, AS, wr * 16, 0);
    acc0 = wmma(a, fragB(Bsm, BS, 0, wc * 32 + 0), acc0);
    acc1 = wmma(a, fragB(Bsm, BS, 0, wc * 32 + 16), acc1);
  }
  int hf = lane >> 4, lr = lane & 15;
#pragma unroll
  for (int nt = 0; nt < 2; ++nt) {
    v8f acc = nt ? acc1 : acc0;
#pragma unroll
    for (int r = 0; r < 8; ++r) {
      int m = mBase + wr * 16 + r + 8 * hf;
      int e = nBase + wc * 32 + nt * 16 + lr;
      int t = m >> 2, bb = m & 3;
      int chunk = e >> 10, ee = e & 1023;
      int h = ee >> 6, d = ee & 63;
      float v = acc[r];
      if (chunk == 0) {
        Qb[((long)(bb * NH + h) * T_LEN + t) * DH + d] = (__bf16)(v * QSCALE);
      } else if (chunk == 1) {
        Kt[((long)(bb * NH + h) * DH + d) * T_LEN + t] = (__bf16)v;  // transposed
      } else {
        Vb[((long)(bb * NH + h) * T_LEN + t) * DH + d] = (__bf16)v;
      }
    }
  }
}

// ---------------- Kernel 2: qp = Q @ pos_emb^T ----------------
// qp[row][j] = sum_d Qb[row][d] * pos_emb[j][d]; rows=B*H*T=65536, j<80
__global__ __launch_bounds__(256) void k_qp(const __bf16* __restrict__ Qb,
                                            const float* __restrict__ pos_emb,
                                            float* __restrict__ qp) {
  const int PJS = 84;
  __shared__ __bf16 posT[64 * 84];  // [d][j]
  int tid = threadIdx.x, lane = tid & 31, wv = tid >> 5;
#pragma unroll
  for (int i = 0; i < 20; ++i) {
    int e = tid + 256 * i;  // 0..5119
    int j = e >> 6, d = e & 63;
    posT[d * PJS + j] = (__bf16)pos_emb[j * DH + d];
  }
  __syncthreads();
  int hf = lane >> 4, lr = lane & 15;
  int m0 = blockIdx.x * 128 + wv * 16;
  v16bf a0 = fragA(Qb, DH, m0, 0);
  v16bf a1 = fragA(Qb, DH, m0, 32);
#pragma unroll
  for (int jt = 0; jt < 5; ++jt) {
    v8f acc = {};
    acc = wmma(a0, fragB(posT, PJS, 0, jt * 16), acc);
    acc = wmma(a1, fragB(posT, PJS, 32, jt * 16), acc);
#pragma unroll
    for (int r = 0; r < 8; ++r) {
      int row = m0 + r + 8 * hf;
      int col = jt * 16 + lr;
      qp[(long)row * 80 + col] = acc[r];
    }
  }
}

// ---------------- Kernel 3: fused flash attention with rel-pos ----------------
// grid (T/128, B*H); block 256 = 8 waves, each wave owns 16 q-rows.
__global__ __launch_bounds__(256) void k_attn(const __bf16* __restrict__ Qb,
                                              const __bf16* __restrict__ Kt,
                                              const __bf16* __restrict__ Vb,
                                              const float* __restrict__ qp,
                                              const float* __restrict__ pos_emb,
                                              __bf16* __restrict__ Cb) {
  const int KS = 72, VS = 72, PS = 68;
  __shared__ alignas(16) char smem[52736];
  int tid = threadIdx.x, lane = tid & 31, wv = tid >> 5;
  int hf = lane >> 4, lr = lane & 15;

  __bf16* Ksm  = (__bf16*)smem;                                  // [64 d][72] = 9216B
  __bf16* Vsm  = (__bf16*)(smem + 9216);                         // [64 t][72] = 9216B
  __bf16* Psm  = (__bf16*)(smem + 18432) + wv * 16 * PS;         // 8 * 2176
  __bf16* Band = (__bf16*)(smem + 35840) + wv * 16 * 64;         // 8 * 2048
  float*  aggSm = (float*)(smem + 52224) + wv * 16;              // 8 * 64
  __bf16* PEsm = (__bf16*)smem;                                  // epilogue alias [96][72]

  unsigned ksmOff = (unsigned)(size_t)Ksm;
  unsigned vsmOff = (unsigned)(size_t)Vsm;

  int bh = blockIdx.y;
  int qBase = blockIdx.x * 128;
  int q0 = qBase + wv * 16;
  const __bf16* Qg  = Qb + (long)bh * T_LEN * DH;
  const __bf16* KTg = Kt + (long)bh * DH * T_LEN;   // [64 d][1024 t]
  const __bf16* Vg  = Vb + (long)bh * T_LEN * DH;
  const float*  qpg = qp + (long)bh * T_LEN * 80;

  v16bf qa0 = fragA(Qg, DH, q0, 0);
  v16bf qa1 = fragA(Qg, DH, q0, 32);

  for (int i = lane; i < 16 * 64; i += 32) Band[i] = (__bf16)0.f;

  float m[8], lsum[8], agg0[8], qp0[8];
  v8f out[4];
#pragma unroll
  for (int r = 0; r < 8; ++r) {
    m[r] = -1e30f; lsum[r] = 0.f; agg0[r] = 0.f;
    qp0[r] = qpg[(q0 + r + 8 * hf) * 80 + 0];
  }
#pragma unroll
  for (int dt = 0; dt < 4; ++dt) out[dt] = (v8f){};

  int nk = qBase / 64 + 2;
  for (int kt = 0; kt < nk; ++kt) {
    int kB = kt * 64;
    __syncthreads();
    // async tile copies: K^T slab [64 d][64 t] and V slab [64 t][64 d], 16B/lane
#pragma unroll
    for (int i = 0; i < 2; ++i) {
      int c = tid + 256 * i;             // 0..511
      int r0 = c >> 3, dp = c & 7;       // row 0..63, 16B chunk 0..7
      async_b128(ksmOff + (unsigned)(r0 * KS + dp * 8) * 2, KTg + r0 * T_LEN + kB + dp * 8);
      async_b128(vsmOff + (unsigned)(r0 * VS + dp * 8) * 2, Vg + (kB + r0) * DH + dp * 8);
    }
    wait_async0();
    __syncthreads();
    bool active = (kB <= q0 + 15);
    if (active) {
      v8f s[4];
#pragma unroll
      for (int nt = 0; nt < 4; ++nt) {
        v8f acc = {};
        acc = wmma(qa0, fragB(Ksm, KS, 0, nt * 16), acc);
        acc = wmma(qa1, fragB(Ksm, KS, 32, nt * 16), acc);
        s[nt] = acc;
      }
      float mt[8];
#pragma unroll
      for (int r = 0; r < 8; ++r) mt[r] = -1e30f;
#pragma unroll
      for (int nt = 0; nt < 4; ++nt)
#pragma unroll
        for (int r = 0; r < 8; ++r) {
          int qrow = q0 + r + 8 * hf;
          int kcol = kB + nt * 16 + lr;
          float sv = s[nt][r];
          int j = kcol - qrow + 64;
          if (j > 64)       sv = -1e30f;           // causal mask (k > q)
          else if (j >= 1)  sv += qpg[qrow * 80 + j];
          else              sv += qp0[r];
          s[nt][r] = sv;
          mt[r] = fmaxf(mt[r], sv);
        }
      float scl[8];
#pragma unroll
      for (int r = 0; r < 8; ++r) {
        float v = mt[r];
        v = fmaxf(v, __shfl_xor(v, 1));
        v = fmaxf(v, __shfl_xor(v, 2));
        v = fmaxf(v, __shfl_xor(v, 4));
        v = fmaxf(v, __shfl_xor(v, 8));
        float mn = fmaxf(m[r], v);
        scl[r] = __expf(m[r] - mn);
        m[r] = mn;
        lsum[r] *= scl[r];
        agg0[r] *= scl[r];
#pragma unroll
        for (int nt = 0; nt < 4; ++nt) out[nt][r] *= scl[r];
      }
      if (kB + 126 >= q0) {  // band region can hold data: keep it scaled to m
        int brow = (lr & 7) + 8 * hf;
        int c0 = ((lr >> 3) & 1) * 32;
        float sc = scl[lr & 7];
        if (sc != 1.f)
          for (int c = 0; c < 32; ++c) {
            float v = (float)Band[brow * 64 + c0 + c];
            Band[brow * 64 + c0 + c] = (__bf16)(v * sc);
          }
      }
#pragma unroll
      for (int nt = 0; nt < 4; ++nt)
#pragma unroll
        for (int r = 0; r < 8; ++r) {
          int qrow = q0 + r + 8 * hf;
          int kcol = kB + nt * 16 + lr;
          float p = __expf(s[nt][r] - m[r]);
          lsum[r] += p;
          int j = kcol - qrow + 64;
          if (j >= 1 && j <= 64) Band[(r + 8 * hf) * 64 + (j - 1)] = (__bf16)p;
          else if (j < 1)        agg0[r] += p;
          Psm[(r + 8 * hf) * PS + nt * 16 + lr] = (__bf16)p;
        }
      v16bf pa0 = fragA(Psm, PS, 0, 0);
      v16bf pa1 = fragA(Psm, PS, 0, 32);
#pragma unroll
      for (int dt = 0; dt < 4; ++dt) {
        out[dt] = wmma(pa0, fragB(Vsm, VS, 0, dt * 16), out[dt]);
        out[dt] = wmma(pa1, fragB(Vsm, VS, 32, dt * 16), out[dt]);
      }
    }
  }
  // lane-partial reductions over the 16-lane (N) groups
#pragma unroll
  for (int r = 0; r < 8; ++r) {
    float v = lsum[r];
    v += __shfl_xor(v, 1); v += __shfl_xor(v, 2);
    v += __shfl_xor(v, 4); v += __shfl_xor(v, 8);
    lsum[r] = v;
    float a = agg0[r];
    a += __shfl_xor(a, 1); a += __shfl_xor(a, 2);
    a += __shfl_xor(a, 4); a += __shfl_xor(a, 8);
    agg0[r] = a;
  }
  if (lane == 0)
    for (int r = 0; r < 8; ++r) aggSm[r] = agg0[r];
  if (lane == 16)
    for (int r = 0; r < 8; ++r) aggSm[8 + r] = agg0[r];
  __syncthreads();
#pragma unroll
  for (int i = 0; i < 24; ++i) {   // pos_emb -> LDS bf16 [96][72] (aliases Ksm/Vsm)
    int e = tid + 256 * i;  // 0..6143
    int j = e >> 6, d = e & 63;
    PEsm[j * 72 + d] = (__bf16)pos_emb[j * DH + d];
  }
  __syncthreads();
  // aggP A-frags: 16 rows x 96 K  (K=0 -> agg0, 1..64 -> band, 65..95 -> 0)
  v16bf ga[3];
#pragma unroll
  for (int c = 0; c < 3; ++c) {
    v16bf g;
#pragma unroll
    for (int i = 0; i < 16; ++i) {
      int kk = c * 32 + (i < 8 ? (hf * 8 + i) : (16 + hf * 8 + (i - 8)));
      float v;
      if (kk == 0)       v = aggSm[lr];
      else if (kk <= 64) v = (float)Band[lr * 64 + (kk - 1)];
      else               v = 0.f;
      g[i] = (__bf16)v;
    }
    ga[c] = g;
  }
#pragma unroll
  for (int dt = 0; dt < 4; ++dt)
#pragma unroll
    for (int c = 0; c < 3; ++c)
      out[dt] = wmma(ga[c], fragB(PEsm, 72, c * 32, dt * 16), out[dt]);
  // normalize + store ctx as bf16 attn_vec[b][t][h*64+d]
  int b = bh >> 4, h = bh & 15;
#pragma unroll
  for (int dt = 0; dt < 4; ++dt)
#pragma unroll
    for (int r = 0; r < 8; ++r) {
      int t = q0 + r + 8 * hf;
      int d = dt * 16 + lr;
      float v = out[dt][r] / lsum[r];
      Cb[((long)(b * T_LEN + t)) * DM + h * 64 + d] = (__bf16)v;
    }
}

// ---------------- Kernel 4: output projection ----------------
// out[t][b][n] = sum_e Cb[b*T+t][e] * Wo[n][e]
__global__ __launch_bounds__(256) void k_out(const __bf16* __restrict__ Cb,
                                             const float* __restrict__ Wo,
                                             float* __restrict__ outp) {
  const int AS = 40, BS = 68;
  __shared__ __bf16 Asm[64 * 40];
  __shared__ __bf16 Bsm[32 * 68];
  int tid = threadIdx.x, lane = tid & 31, wv = tid >> 5;
  int wr = wv >> 1, wc = wv & 1;
  int mBase = blockIdx.x * 64, nBase = blockIdx.y * 64;
  v8f acc0 = {}, acc1 = {};
  for (int kb = 0; kb < DM; kb += 32) {
    __syncthreads();
#pragma unroll
    for (int i = 0; i < 8; ++i) {
      int e = tid + 256 * i;
      int mm = e >> 5, k = e & 31;
      Asm[mm * AS + k] = Cb[(long)(mBase + mm) * DM + kb + k];
    }
#pragma unroll
    for (int i = 0; i < 8; ++i) {
      int e = tid + 256 * i;
      int n = e >> 5, k = e & 31;
      Bsm[k * BS + n] = (__bf16)Wo[(long)(nBase + n) * DM + kb + k];
    }
    if (kb + 32 < DM) {
      __builtin_prefetch(&Wo[(long)(nBase + (tid & 63)) * DM + kb + 32], 0, 3);
    }
    __syncthreads();
    v16bf a = fragA(Asm, AS, wr * 16, 0);
    acc0 = wmma(a, fragB(Bsm, BS, 0, wc * 32 + 0), acc0);
    acc1 = wmma(a, fragB(Bsm, BS, 0, wc * 32 + 16), acc1);
  }
  int hf = lane >> 4, lr = lane & 15;
#pragma unroll
  for (int nt = 0; nt < 2; ++nt) {
    v8f acc = nt ? acc1 : acc0;
#pragma unroll
    for (int r = 0; r < 8; ++r) {
      int mrow = mBase + wr * 16 + r + 8 * hf;  // = b*T + t
      int b = mrow >> 10, t = mrow & 1023;
      int col = nBase + wc * 32 + nt * 16 + lr;
      outp[((long)(t * BATCH + b)) * DM + col] = acc[r];
    }
  }
}

extern "C" void kernel_launch(void* const* d_in, const int* in_sizes, int n_in,
                              void* d_out, int out_size, void* d_ws, size_t ws_size,
                              hipStream_t stream) {
  (void)in_sizes; (void)n_in; (void)out_size; (void)ws_size;
  const float* w        = (const float*)d_in[0];
  // d_in[1] = attn_mask: implicit causal mask recomputed in-kernel, not read
  const float* Wqkv     = (const float*)d_in[2];
  const float* pos_emb  = (const float*)d_in[3];
  const float* Wo       = (const float*)d_in[4];

  char* ws = (char*)d_ws;
  __bf16* Qb = (__bf16*)(ws);                 //  8 MB  [B,H,T,64]
  __bf16* Kt = (__bf16*)(ws + (8l << 20));    //  8 MB  [B,H,64,T]  (pre-transposed)
  __bf16* Vb = (__bf16*)(ws + (16l << 20));   //  8 MB  [B,H,T,64]
  float*  qp = (float*) (ws + (24l << 20));   // 20 MB  [B*H*T, 80]
  __bf16* Cb = (__bf16*)(ws + (44l << 20));   //  8 MB  [B,T,1024]

  dim3 blk(256);
  k_qkv<<<dim3(64, 48), blk, 0, stream>>>(w, Wqkv, Qb, Kt, Vb);
  k_qp <<<dim3(512),    blk, 0, stream>>>(Qb, pos_emb, qp);
  k_attn<<<dim3(8, 64), blk, 0, stream>>>(Qb, Kt, Vb, qp, pos_emb, Cb);
  k_out<<<dim3(64, 16), blk, 0, stream>>>(Cb, Wo, (float*)d_out);
}
